// ElectronicSchnet_2293512536106
// MI455X (gfx1250) — matrix operand order repaired and checked by reading, not compiled, not run on hardware
//
#include <hip/hip_runtime.h>

typedef __attribute__((ext_vector_type(16))) _Float16 v16h;
typedef __attribute__((ext_vector_type(8)))  float    v8f;

#define NBATCH 1024
#define NELEC  32
#define EMBED  128
#define BASIS  32
#define KERN   64
#define NNUC   8
#define HID    45
#define HIDP   48

// packed-weight layout (in _Float16 units); all matrices stored FRAGMENT-MAJOR:
//   frag[tile][lane][t] with tile = kt*ntiles + nt, t = 0..15,
//   element = B[kt*32 + (lane>>4)*16 + t][nt*16 + (lane&15)]
#define WP_WH   0          // Wh   K=128,N=64  -> 16 tiles (ntiles=4)
#define WP_G    8192       // Gcat K=192,N=128 -> 48 tiles (ntiles=8) rows: GT|GF|GN
#define WP_MLP  32768      // per-label: W1 frags (3 tiles, 1536) + W2 frags (8 tiles, 4096)
#define WP_LBL  5632
#define WP_TOTAL (WP_MLP + 3 * WP_LBL)   // 49664 halves

static __device__ __forceinline__ float sspf(float v) {
  float sp = (v > 20.0f) ? v : __logf(1.0f + __expf(v));
  return sp - 0.69314718055994531f;   // softplus(v) - log(2)
}

// ---- WMMA fragment loaders (contiguous k per lane half, same order as packed B, wave32) ----
// A (16x32 f16): lane row = lane&15; halves 0..15 = K (lanehi?16..31:0..15)
static __device__ __forceinline__ v16h load_a(const float* row, int kbase, int lanehi) {
  v16h a;
  int k0 = kbase + (lanehi ? 16 : 0);
  int k1 = k0 + 8;
#pragma unroll
  for (int t = 0; t < 8; ++t) { a[t] = (_Float16)row[k0 + t]; a[t + 8] = (_Float16)row[k1 + t]; }
  return a;
}
static __device__ __forceinline__ v16h load_a(const _Float16* row, int kbase, int lanehi) {
  v16h a;
  int k0 = kbase + (lanehi ? 16 : 0);
  int k1 = k0 + 8;
#pragma unroll
  for (int t = 0; t < 8; ++t) { a[t] = row[k0 + t]; a[t + 8] = row[k1 + t]; }
  return a;
}
// B fragment: one contiguous, 32B-aligned v16h per (tile, lane)
static __device__ __forceinline__ v16h load_b_frag(const _Float16* base, int tile, int lane) {
  return *(const v16h*)(base + (((size_t)tile << 5) + lane) * 16);
}

// spin-pair index arithmetic (pairs are receiver-major per block)
static __device__ __forceinline__ void pair_ij(int blk, int p, int& i, int& j) {
  if (blk == 0)      { i = p / 15; int jj = p % 15; j = jj + (jj >= i ? 1 : 0); }          // uu
  else if (blk == 1) { i = p >> 4; j = 16 + (p & 15); }                                    // ud
  else if (blk == 2) { i = 16 + (p >> 4); j = p & 15; }                                    // du
  else               { int ii = p / 15; int jj = p % 15; i = 16 + ii; j = 16 + jj + (jj >= ii ? 1 : 0); } // dd
}

// ---------------------------------------------------------------------------
__global__ void init_x_kernel(const float* __restrict__ X, float* __restrict__ x) {
  int gid = blockIdx.x * blockDim.x + threadIdx.x;
  x[gid] = X[gid & (EMBED - 1)];
}

__global__ void cvt_f16_kernel(const float* __restrict__ src, _Float16* __restrict__ dst, int n) {
  int stride = gridDim.x * blockDim.x;
  for (int i = blockIdx.x * blockDim.x + threadIdx.x; i < n; i += stride)
    dst[i] = (_Float16)src[i];
}

// ---------------------------------------------------------------------------
static __device__ __forceinline__ void pack_frag(const float* __restrict__ src, int Ksrc, int Nsrc,
                                                 int Kp, int Np, _Float16* __restrict__ dst,
                                                 int tid, int stride) {
  int ntiles = Np >> 4;
  int total = Kp * Np;
  for (int idx = tid; idx < total; idx += stride) {
    int t = idx & 15, lane = (idx >> 4) & 31, tile = idx >> 9;
    int kt = tile / ntiles, nt = tile % ntiles;
    int k = kt * 32 + (lane >> 4) * 16 + t;
    int n = nt * 16 + (lane & 15);
    float v = (k < Ksrc && n < Nsrc) ? src[k * Nsrc + n] : 0.0f;
    dst[idx] = (_Float16)v;
  }
}

__global__ void pack_weights(const float* __restrict__ Wh,
                             const float* __restrict__ GT, const float* __restrict__ GF,
                             const float* __restrict__ GN,
                             const float* __restrict__ W1T, const float* __restrict__ W2T,
                             const float* __restrict__ W1F, const float* __restrict__ W2F,
                             const float* __restrict__ W1N, const float* __restrict__ W2N,
                             _Float16* __restrict__ wp) {
  int tid = blockIdx.x * blockDim.x + threadIdx.x;
  int stride = gridDim.x * blockDim.x;
  pack_frag(Wh, 128, 64, 128, 64, wp + WP_WH, tid, stride);
  // Gcat (K rows 0..63 = GT, 64..127 = GF, 128..191 = GN), fragment-major
  for (int idx = tid; idx < 192 * 128; idx += stride) {
    int t = idx & 15, lane = (idx >> 4) & 31, tile = idx >> 9;
    int kt = tile >> 3, nt = tile & 7;
    int k = kt * 32 + (lane >> 4) * 16 + t;
    int n = nt * 16 + (lane & 15);
    float v = (k < 64) ? GT[k * 128 + n] : (k < 128) ? GF[(k - 64) * 128 + n] : GN[(k - 128) * 128 + n];
    wp[WP_G + idx] = (_Float16)v;
  }
  const float* W1s[3] = {W1T, W1F, W1N};
  const float* W2s[3] = {W2T, W2F, W2N};
#pragma unroll
  for (int l = 0; l < 3; ++l) {
    _Float16* base = wp + WP_MLP + l * WP_LBL;
    pack_frag(W1s[l], 32, HID, 32, HIDP, base, tid, stride);          // 3 tiles
    pack_frag(W2s[l], HID, 64, 64, 64, base + 1536, tid, stride);     // 8 tiles (K rows >=45 zero)
  }
}

// ---------------------------------------------------------------------------
// h = x @ Wh : M = NBATCH*32, K = 128, N = 64. One 16-row tile per block, 4 waves = 4 N-tiles.
__global__ __launch_bounds__(128) void h_gemm(const float* __restrict__ x,
                                              const _Float16* __restrict__ WhF,
                                              float* __restrict__ hbuf) {
  int mt = blockIdx.x;
  int wave = threadIdx.x >> 5, lane = threadIdx.x & 31;
  int lanehi = lane >> 4, lanelo = lane & 15;
  const float* arow = x + ((size_t)mt * 16 + lanelo) * EMBED;
  v8f c = {};
#pragma unroll
  for (int kk = 0; kk < 4; ++kk) {
    v16h a = load_a(arow, kk * 32, lanehi);
    v16h b = load_b_frag(WhF, kk * 4 + wave, lane);
    c = __builtin_amdgcn_wmma_f32_16x16x32_f16(false, a, false, b, (short)0, c, false, false);
  }
  float* out = hbuf + (size_t)mt * 16 * KERN + wave * 16 + lanelo;
#pragma unroll
  for (int r = 0; r < 8; ++r) out[(size_t)(r + (lanehi ? 8 : 0)) * KERN] = c[r];
}

// ---------------------------------------------------------------------------
// One workgroup per batch element: edge MLPs (elec + nuc) + gated segment-sum into z[32][192].
template <typename T>
__global__ __launch_bounds__(256) void edge_kernel(
    const T* __restrict__ edges_elec, const T* __restrict__ edges_nuc,
    const float* __restrict__ hbuf, const _Float16* __restrict__ wp,
    const float* __restrict__ b1T, const float* __restrict__ b2T,
    const float* __restrict__ b1F, const float* __restrict__ b2F,
    const float* __restrict__ b1N, const float* __restrict__ b2N,
    const float* __restrict__ Y, _Float16* __restrict__ zbuf) {
  __shared__ float h_s[NELEC][KERN];
  __shared__ float z_s[NELEC][192];
  __shared__ float b1_s[3][HIDP];
  __shared__ float b2_s[3][KERN];
  __shared__ float Y_s[NNUC][KERN];
  __shared__ _Float16 stage[8][16][64];

  int b = blockIdx.x;
  int tid = threadIdx.x;
  int wave = tid >> 5, lane = tid & 31, lanehi = lane >> 4, lanelo = lane & 15;

  for (int i = tid; i < NELEC * KERN; i += 256) ((float*)h_s)[i] = hbuf[(size_t)b * NELEC * KERN + i];
  for (int i = tid; i < NELEC * 192; i += 256) ((float*)z_s)[i] = 0.0f;
  for (int i = tid; i < 3 * HIDP; i += 256) {
    int l = i / HIDP, c = i % HIDP;
    const float* src = (l == 0) ? b1T : (l == 1) ? b1F : b1N;
    b1_s[l][c] = (c < HID) ? src[c] : 0.0f;   // pad bias 0 -> ssp(0) == 0
  }
  for (int i = tid; i < 3 * KERN; i += 256) {
    int l = i / KERN, c = i & (KERN - 1);
    const float* src = (l == 0) ? b2T : (l == 1) ? b2F : b2N;
    b2_s[l][c] = src[c];
  }
  for (int i = tid; i < NNUC * KERN; i += 256) ((float*)Y_s)[i] = Y[i];
  for (int i = lane; i < 256; i += 32) stage[wave][i >> 4][48 + (i & 15)] = (_Float16)0.0f; // K pad
  __syncthreads();

  // 78 sixteen-row jobs: 15 uu, 16 ud, 16 du, 15 dd, 16 nuclei tiles.
  for (int job = wave; job < 78; job += 8) {
    int blk, t;
    if (job < 15)      { blk = 0; t = job; }
    else if (job < 31) { blk = 1; t = job - 15; }
    else if (job < 47) { blk = 2; t = job - 31; }
    else if (job < 62) { blk = 3; t = job - 47; }
    else               { blk = 4; t = job - 62; }
    int lbl = (blk == 0 || blk == 3) ? 0 : ((blk == 4) ? 2 : 1);
    int t16 = t * 16;
    const _Float16* W1f = wp + WP_MLP + lbl * WP_LBL;   // 3 fragment tiles
    const _Float16* W2f = W1f + 1536;                   // 8 fragment tiles

    // input edge rows -> A fragment (16x32)
    const T* arow;
    if (blk == 4) {
      int rr = t16 + lanelo;
      arow = edges_nuc + ((size_t)b * (NELEC * NNUC) + rr) * BASIS;
    } else {
      int ii, jj; pair_ij(blk, t16 + lanelo, ii, jj);
      arow = edges_elec + (((size_t)b * NELEC + ii) * NELEC + jj) * BASIS;
    }
    v16h a0 = load_a(arow, 0, lanehi);

    // GEMM1: [16x32] @ [32x48]
    v8f c1[3];
#pragma unroll
    for (int nt = 0; nt < 3; ++nt) {
      v16h bf = load_b_frag(W1f, nt, lane);
      v8f z = {};
      c1[nt] = __builtin_amdgcn_wmma_f32_16x16x32_f16(false, a0, false, bf, (short)0, z, false, false);
    }
    // bias + shifted-softplus, restage as f16 A matrix for GEMM2
#pragma unroll
    for (int nt = 0; nt < 3; ++nt) {
      int col = nt * 16 + lanelo;
      float bb = b1_s[lbl][col];
#pragma unroll
      for (int r = 0; r < 8; ++r)
        stage[wave][r + (lanehi ? 8 : 0)][col] = (_Float16)sspf(c1[nt][r] + bb);
    }
    asm volatile("s_wait_dscnt 0" ::: "memory");   // in-wave LDS store->load ordering

    // GEMM2: [16x64] @ [64x64]
    v8f c2[4] = {v8f{}, v8f{}, v8f{}, v8f{}};
#pragma unroll
    for (int kk = 0; kk < 2; ++kk) {
      v16h a1 = load_a(&stage[wave][lanelo][0], kk * 32, lanehi);
#pragma unroll
      for (int nt = 0; nt < 4; ++nt) {
        v16h bf = load_b_frag(W2f, kk * 4 + nt, lane);
        c2[nt] = __builtin_amdgcn_wmma_f32_16x16x32_f16(false, a1, false, bf, (short)0, c2[nt], false, false);
      }
    }

    // bias2, gate with h[j] (or Y[nuc]), segment-sum per receiver
#pragma unroll
    for (int nt = 0; nt < 4; ++nt) {
      int col = nt * 16 + lanelo;
      float bb = b2_s[lbl][col];
#pragma unroll
      for (int r = 0; r < 8; ++r) {
        int pr = t16 + r + (lanehi ? 8 : 0);
        int recv, zc; float mult;
        if (blk == 4) { recv = pr >> 3; zc = 128; mult = Y_s[pr & 7][col]; }
        else { int ii, jj; pair_ij(blk, pr, ii, jj); recv = ii; zc = (lbl == 0) ? 0 : 64; mult = h_s[jj][col]; }
        atomicAdd(&z_s[recv][zc + col], (c2[nt][r] + bb) * mult);
      }
    }
  }
  __syncthreads();
  for (int i = tid; i < NELEC * 192; i += 256)
    zbuf[(size_t)b * NELEC * 192 + i] = (_Float16)((float*)z_s)[i];
}

// ---------------------------------------------------------------------------
// x += z @ Gcat : M = NBATCH*32, K = 192, N = 128. 8 waves = 8 N-tiles per 16-row tile.
__global__ __launch_bounds__(256) void g_gemm(const _Float16* __restrict__ zbuf,
                                              const _Float16* __restrict__ GcatF,
                                              float* __restrict__ x) {
  int mt = blockIdx.x;
  int wave = threadIdx.x >> 5, lane = threadIdx.x & 31;
  int lanehi = lane >> 4, lanelo = lane & 15;
  const _Float16* arow = zbuf + ((size_t)mt * 16 + lanelo) * 192;
  v8f c = {};
#pragma unroll
  for (int kk = 0; kk < 6; ++kk) {
    v16h a = load_a(arow, kk * 32, lanehi);
    v16h b = load_b_frag(GcatF, kk * 8 + wave, lane);
    c = __builtin_amdgcn_wmma_f32_16x16x32_f16(false, a, false, b, (short)0, c, false, false);
  }
  float* out = x + (size_t)mt * 16 * EMBED + wave * 16 + lanelo;
#pragma unroll
  for (int r = 0; r < 8; ++r) out[(size_t)(r + (lanehi ? 8 : 0)) * EMBED] += c[r];
}

// ---------------------------------------------------------------------------
extern "C" void kernel_launch(void* const* d_in, const int* in_sizes, int n_in,
                              void* d_out, int out_size, void* d_ws, size_t ws_size,
                              hipStream_t stream) {
  (void)out_size;
  const float* edges_elec = (const float*)d_in[0];
  const float* edges_nuc  = (const float*)d_in[1];
  const float *Xp = nullptr, *Yp = nullptr;
  const float *G[3][3], *Hm[3], *W1[3][3], *B1[3][3], *W2[3][3], *B2[3][3]; // [iter][lbl: 0=True,1=False,2=n]

  if (n_in >= 3 && in_sizes[2] == 128) {
    // jax pytree (sorted keys): X, Y, g(9: False,True,n), h(3), w(9 x {W1,b1,W2,b2})
    Xp = (const float*)d_in[2];
    Yp = (const float*)d_in[3];
    const int lblmap[3] = {1, 0, 2};   // storage order False,True,n -> label index
    for (int n = 0; n < 3; ++n) {
      for (int s = 0; s < 3; ++s) {
        int l = lblmap[s];
        G[n][l]  = (const float*)d_in[4 + n * 3 + s];
        int base = 16 + (n * 3 + s) * 4;
        W1[n][l] = (const float*)d_in[base + 0];
        B1[n][l] = (const float*)d_in[base + 1];
        W2[n][l] = (const float*)d_in[base + 2];
        B2[n][l] = (const float*)d_in[base + 3];
      }
      Hm[n] = (const float*)d_in[13 + n];
    }
  } else {
    // insertion order: Y, X, w(9: True,False,n), h(3), g(9)
    Yp = (const float*)d_in[2];
    Xp = (const float*)d_in[3];
    for (int n = 0; n < 3; ++n) {
      for (int s = 0; s < 3; ++s) {      // storage order == label order True,False,n
        int base = 4 + (n * 3 + s) * 4;
        W1[n][s] = (const float*)d_in[base + 0];
        B1[n][s] = (const float*)d_in[base + 1];
        W2[n][s] = (const float*)d_in[base + 2];
        B2[n][s] = (const float*)d_in[base + 3];
        G[n][s]  = (const float*)d_in[43 + n * 3 + s];
      }
      Hm[n] = (const float*)d_in[40 + n];
    }
  }

  const size_t NE_ELEC = (size_t)NBATCH * NELEC * NELEC * BASIS;   // 33,554,432
  const size_t NE_NUC  = (size_t)NBATCH * NELEC * NNUC * BASIS;    //  8,388,608

  float* x       = (float*)d_out;                                  // [NBATCH,32,128]
  float* hbuf    = (float*)d_ws;                                   // [NBATCH,32,64] f32
  _Float16* zbuf = (_Float16*)(hbuf + (size_t)NBATCH * NELEC * KERN); // [NBATCH,32,192] f16
  _Float16* wp   = zbuf + (size_t)NBATCH * NELEC * 192;            // packed fragments
  _Float16* e16e = wp + WP_TOTAL;                                  // optional f16 edge copies
  _Float16* e16n = e16e + NE_ELEC;

  size_t base_need = (size_t)NBATCH * NELEC * KERN * 4 +
                     (size_t)NBATCH * NELEC * 192 * 2 + (size_t)WP_TOTAL * 2;
  bool use_f16_edges = ws_size >= base_need + (NE_ELEC + NE_NUC) * 2;

  init_x_kernel<<<(NBATCH * NELEC * EMBED) / 256, 256, 0, stream>>>(Xp, x);
  if (use_f16_edges) {
    cvt_f16_kernel<<<8192, 256, 0, stream>>>(edges_elec, e16e, (int)NE_ELEC);
    cvt_f16_kernel<<<4096, 256, 0, stream>>>(edges_nuc, e16n, (int)NE_NUC);
  }

  for (int n = 0; n < 3; ++n) {
    pack_weights<<<64, 256, 0, stream>>>(Hm[n], G[n][0], G[n][1], G[n][2],
                                         W1[n][0], W2[n][0], W1[n][1], W2[n][1],
                                         W1[n][2], W2[n][2], wp);
    h_gemm<<<(NBATCH * NELEC) / 16, 128, 0, stream>>>(x, wp + WP_WH, hbuf);
    if (use_f16_edges) {
      edge_kernel<_Float16><<<NBATCH, 256, 0, stream>>>(e16e, e16n, hbuf, wp,
                                                        B1[n][0], B2[n][0], B1[n][1], B2[n][1],
                                                        B1[n][2], B2[n][2], Yp, zbuf);
    } else {
      edge_kernel<float><<<NBATCH, 256, 0, stream>>>(edges_elec, edges_nuc, hbuf, wp,
                                                     B1[n][0], B2[n][0], B1[n][1], B2[n][1],
                                                     B1[n][2], B2[n][2], Yp, zbuf);
    }
    g_gemm<<<(NBATCH * NELEC) / 16, 256, 0, stream>>>(zbuf, wp + WP_G, x);
  }
}
